// SS3D_Block_88467736363574
// MI455X (gfx1250) — matrix operand order repaired and checked by reading, not compiled, not run on hardware
//
#include <hip/hip_runtime.h>
#include <hip/hip_bf16.h>

typedef __attribute__((ext_vector_type(16))) _Float16 v16h;
typedef __attribute__((ext_vector_type(8)))  float    v8f;

constexpr int NB     = 2;        // batch
constexpr int SEQL   = 1024;     // L = 32*32
constexpr int DMODEL = 192;
constexpr int DINNER = 384;
constexpr int E2     = 768;      // 2*DINNER
constexpr int NSTATE = 16;
constexpr int DTRANK = 12;
constexpr int CINNER = 16;
constexpr int TCHAN  = 24576;    // channel-scan length = 768*32
constexpr int PERSB  = 393216;   // DINNER*SEQL == CINNER*TCHAN
constexpr int NPROJ  = 44;       // DTRANK + 2*NSTATE
constexpr int NPADR  = 48;       // padded to 3 WMMA M-tiles

// chunked-scan geometry
constexpr int CS_SEQ = 128, PC_SEQ = 8;    // 8 chunks of 128 (seq scans)
constexpr int CS_CH  = 256, PC_CH  = 96;   // 96 chunks of 256 (channel scans)
constexpr int CS_SEQ_TOTAL = NB * 2 * DINNER * NSTATE * PC_SEQ;   // 196608
constexpr int CS_CH_TOTAL  = NB * 4 * CINNER * NSTATE * PC_CH;    // 196608

__device__ inline float softplusf(float x) {
  return (x > 20.f) ? x : log1pf(__expf(x));
}
__device__ inline float siluf(float x) {
  return x / (1.f + __expf(-x));
}

// Flat index into xz (b, 768, 1024) for scan s, logical channel ch, position t.
// s0: row-major fwd; s1: rev; s2/s3: ch_h fwd/rev; s4/s5: ch_w fwd/rev.
__device__ inline size_t xz_elem(int s, int b, int ch, int t, int T) {
  int tt = (s & 1) ? (T - 1 - t) : t;
  int e, l;
  if (s < 2)      { e = ch;      l = tt; }
  else if (s < 4) { e = tt >> 5; l = (ch << 5) + (tt & 31); }
  else            { e = tt >> 5; l = ((tt & 31) << 5) + ch; }
  return (size_t)b * (E2 * SEQL) + (size_t)e * SEQL + (size_t)l;
}

// ---------------- K1: f32 -> f16 prep (+ zero-padded xproj weights) ----------
__global__ void k_prep(const float* __restrict__ hidden, const float* __restrict__ w_in,
                       const float* __restrict__ w_out, const float* __restrict__ xproj_s,
                       const float* __restrict__ xproj_c,
                       _Float16* __restrict__ hidden_h, _Float16* __restrict__ w_in_h,
                       _Float16* __restrict__ w_out_h, _Float16* __restrict__ xproj_h) {
  const int N1 = NB * SEQL * DMODEL;   // 393216
  const int N2 = E2 * DMODEL;          // 147456
  const int N3 = DMODEL * DINNER;      // 73728
  int id = blockIdx.x * blockDim.x + threadIdx.x;     // exactly 724992 launched
  if (id < N1) {
    hidden_h[id] = (_Float16)hidden[id];
  } else if (id < N1 + N2) {
    int q = id - N1; w_in_h[q] = (_Float16)w_in[q];
  } else if (id < N1 + N2 + N3) {
    int q = id - N1 - N2; w_out_h[q] = (_Float16)w_out[q];
  } else {
    int q = id - N1 - N2 - N3;                   // 6*48*384
    int s = q / (NPADR * DINNER);
    int r = q % (NPADR * DINNER);
    int row = r / DINNER, col = r % DINNER;
    float v = 0.f;
    if (row < NPROJ) {
      if (s < 2) v = xproj_s[((size_t)s * NPROJ + row) * DINNER + col];
      else if (col < CINNER) v = xproj_c[((size_t)(s - 2) * NPROJ + row) * CINNER + col];
    }
    xproj_h[q] = (_Float16)v;
  }
}

// ---------------- K2: xz[b,e,l] = W_in(768x192) @ H^T via WMMA ---------------
__global__ void k_in_proj(const _Float16* __restrict__ w_in_h,
                          const _Float16* __restrict__ hidden_h,
                          float* __restrict__ xz) {
  int lane = threadIdx.x & 31, wid = threadIdx.x >> 5;
  int tile = blockIdx.x * 4 + wid;            // 6144 tiles = 2 * 48 * 64
  int b = tile / 3072, r = tile % 3072;
  int e0 = (r / 64) * 16, l0 = (r % 64) * 16;
  bool hi = lane >= 16;
  int mr = lane & 15;
  v8f acc = {};
  for (int k0 = 0; k0 < DMODEL; k0 += 32) {
    v16h a, bm;
#pragma unroll
    for (int v = 0; v < 8; ++v)
#pragma unroll
      for (int j = 0; j < 2; ++j) {
        int k = ((v < 4) ? (2 * v + j) : (16 + 2 * (v - 4) + j)) + (hi ? 8 : 0);
        a[2 * v + j] = w_in_h[(size_t)(e0 + mr) * DMODEL + (k0 + k)];
      }
    int kb = hi ? 8 : 0;
#pragma unroll
    for (int i = 0; i < 16; ++i)
      bm[i] = hidden_h[((size_t)b * SEQL + (l0 + mr)) * DMODEL + (k0 + kb + (i < 8 ? i : i + 8))];
    acc = __builtin_amdgcn_wmma_f32_16x16x32_f16(false, a, false, bm, (short)0, acc, false, false);
  }
#pragma unroll
  for (int v = 0; v < 8; ++v) {
    int m = v + (hi ? 8 : 0);
    xz[(size_t)b * E2 * SEQL + (size_t)(e0 + m) * SEQL + (l0 + mr)] = acc[v];
  }
}

// ---------------- K3: causal depthwise conv(K=4) + SiLU, all 6 scans ---------
__global__ void k_conv_silu(const float* __restrict__ xz,
                            const float* __restrict__ conv_w_s, const float* __restrict__ conv_b_s,
                            const float* __restrict__ conv_w_c, const float* __restrict__ conv_b_c,
                            _Float16* __restrict__ xconv) {
  int id = blockIdx.x * blockDim.x + threadIdx.x;   // exactly 6*NB*PERSB launched
  int sb = id / PERSB, s = sb / NB, b = sb % NB;
  int rem = id % PERSB;
  int T = (s < 2) ? SEQL : TCHAN;
  int c = rem / T, t = rem % T;
  const float* wv; float bias;
  if (s < 2) { wv = conv_w_s + ((size_t)s * DINNER + c) * 4;       bias = conv_b_s[s * DINNER + c]; }
  else       { wv = conv_w_c + ((size_t)(s - 2) * CINNER + c) * 4; bias = conv_b_c[(s - 2) * CINNER + c]; }
  float acc = bias;
#pragma unroll
  for (int k = 0; k < 4; ++k) {
    int tp = t - 3 + k;
    if (tp >= 0) acc += wv[k] * xz[xz_elem(s, b, c, tp, T)];
  }
  xconv[id] = (_Float16)siluf(acc);
}

// ---------------- K4: x_dbl = Wproj_pad(48xC) @ xconv(CxT) via WMMA ----------
__global__ void k_xdbl(const _Float16* __restrict__ xconv,
                       const _Float16* __restrict__ xproj_h,
                       _Float16* __restrict__ xdbl) {
  int lane = threadIdx.x & 31, wid = threadIdx.x >> 5;
  int tile = blockIdx.x * 4 + wid;             // 37632 tiles total
  int s, b, mt, nt, T, C;
  if (tile < 768) {                            // seq scans: 2*2*3*64
    s = tile / 384; int r = tile % 384; b = r / 192; int r2 = r % 192;
    mt = r2 / 64; nt = r2 % 64; T = SEQL; C = DINNER;
  } else {                                     // channel scans: 4*2*3*1536
    int t2 = tile - 768; s = 2 + t2 / 9216; int r = t2 % 9216; b = r / 4608;
    int r2 = r % 4608; mt = r2 / 1536; nt = r2 % 1536; T = TCHAN; C = CINNER;
  }
  const _Float16* xc = xconv + (size_t)(s * NB + b) * PERSB;
  const _Float16* Wp = xproj_h + (size_t)s * NPADR * DINNER;
  size_t dbase = (s < 2) ? (size_t)(s * NB + b) * NPADR * SEQL
                         : (size_t)4 * NPADR * SEQL + (size_t)((s - 2) * NB + b) * (size_t)NPADR * TCHAN;
  bool hi = lane >= 16;
  int mr = lane & 15;
  v8f acc = {};
  for (int k0 = 0; k0 < C; k0 += 32) {
    v16h a, bm;
#pragma unroll
    for (int v = 0; v < 8; ++v)
#pragma unroll
      for (int j = 0; j < 2; ++j) {
        int k = ((v < 4) ? (2 * v + j) : (16 + 2 * (v - 4) + j)) + (hi ? 8 : 0);
        a[2 * v + j] = Wp[(size_t)(mt * 16 + mr) * DINNER + (k0 + k)];
      }
    int kb = hi ? 8 : 0;
#pragma unroll
    for (int i = 0; i < 16; ++i) {
      int kk = k0 + kb + (i < 8 ? i : i + 8);
      bm[i] = (kk < C) ? xc[(size_t)kk * T + (nt * 16 + mr)] : (_Float16)0.f;
    }
    acc = __builtin_amdgcn_wmma_f32_16x16x32_f16(false, a, false, bm, (short)0, acc, false, false);
  }
#pragma unroll
  for (int v = 0; v < 8; ++v) {
    int m = v + (hi ? 8 : 0);
    xdbl[dbase + (size_t)(mt * 16 + m) * T + (nt * 16 + mr)] = (_Float16)acc[v];
  }
}

// ------- K5a: sequence scans, chunked (phase 0: states, phase 1: outputs) ----
// grid 192 = s(2) * b(2) * chunk(8) * dgroup(6); block 1024 = dlocal(64) * n(16)
__global__ void k_scan_seq2(int phase,
    const float* __restrict__ xz, const _Float16* __restrict__ xconv,
    const _Float16* __restrict__ xdbl,
    const float* __restrict__ dt_w_s, const float* __restrict__ dt_b_s,
    const float* __restrict__ A_log_s, const float* __restrict__ D_s,
    float* __restrict__ cs_hend, float* __restrict__ cs_aprod,
    const float* __restrict__ cs_hinit, float* __restrict__ ybuf) {
  int dg    = blockIdx.x % 6;
  int chunk = (blockIdx.x / 6) % PC_SEQ;
  int b     = (blockIdx.x / (6 * PC_SEQ)) % NB;
  int s     = blockIdx.x / (6 * PC_SEQ * NB);
  int n  = threadIdx.x & 15;
  int d  = dg * 64 + (threadIdx.x >> 4);
  int sb = s * NB + b;
  __shared__ float sh[NPROJ];
  size_t xcb = (size_t)sb * PERSB + (size_t)d * SEQL;
  size_t dbb = (size_t)sb * NPADR * SEQL;
  float Av = -__expf(A_log_s[(s * DINNER + d) * NSTATE + n]);
  float dtw[DTRANK];
#pragma unroll
  for (int r = 0; r < DTRANK; ++r) dtw[r] = dt_w_s[(s * DINNER + d) * DTRANK + r];
  float dtb = dt_b_s[s * DINNER + d];
  float Dp  = D_s[s * DINNER + d];
  size_t ci = (((size_t)sb * DINNER + d) * NSTATE + n) * PC_SEQ + chunk;
  float h = phase ? cs_hinit[ci] : 0.f;
  float aprod = 1.f;
  int t0 = chunk * CS_SEQ;
  for (int tt = 0; tt < CS_SEQ; ++tt) {
    int t = t0 + tt;
    if (threadIdx.x < NPROJ) sh[threadIdx.x] = (float)xdbl[dbb + (size_t)threadIdx.x * SEQL + t];
    __syncthreads();
    float x = (float)xconv[xcb + t];
    if (tt + 1 < CS_SEQ) __builtin_prefetch(&xconv[xcb + t + 1], 0, 1);
    float pre = dtb;
#pragma unroll
    for (int r = 0; r < DTRANK; ++r) pre += dtw[r] * sh[r];
    float dt = softplusf(pre);
    float a = __expf(dt * Av);
    h = a * h + dt * sh[DTRANK + n] * x;
    if (phase) {
      float yp = h * sh[DTRANK + NSTATE + n];
      yp += __shfl_xor(yp, 8, 16);
      yp += __shfl_xor(yp, 4, 16);
      yp += __shfl_xor(yp, 2, 16);
      yp += __shfl_xor(yp, 1, 16);
      if (n == 0) {
        float z = xz[xz_elem(s, b, DINNER + d, t, SEQL)];
        ybuf[xcb + t] = (yp + Dp * x) * siluf(z);
      }
    } else {
      aprod *= a;
    }
    __syncthreads();
  }
  if (!phase) { cs_hend[ci] = h; cs_aprod[ci] = aprod; }
}

// ------- K5b: channel scans, chunked; grid 768 = sc(4)*b(2)*chunk(96) --------
// block 256 = d(16) * n(16)
__global__ void k_scan_ch2(int phase,
    const float* __restrict__ xz, const _Float16* __restrict__ xconv,
    const _Float16* __restrict__ xdbl,
    const float* __restrict__ dt_w_c, const float* __restrict__ dt_b_c,
    const float* __restrict__ A_log_c, const float* __restrict__ D_c,
    float* __restrict__ cs_hend, float* __restrict__ cs_aprod,
    const float* __restrict__ cs_hinit, float* __restrict__ ybuf) {
  int chunk = blockIdx.x % PC_CH;
  int b     = (blockIdx.x / PC_CH) % NB;
  int sc    = blockIdx.x / (PC_CH * NB);
  int s     = 2 + sc;
  int n = threadIdx.x & 15, d = threadIdx.x >> 4;
  int sb = sc * NB + b;
  __shared__ float sh[NPROJ];
  size_t xcb = (size_t)(s * NB + b) * PERSB + (size_t)d * TCHAN;
  size_t dbb = (size_t)4 * NPADR * SEQL + (size_t)sb * (size_t)NPADR * TCHAN;
  float Av = -__expf(A_log_c[(sc * CINNER + d) * NSTATE + n]);
  float dtw[DTRANK];
#pragma unroll
  for (int r = 0; r < DTRANK; ++r) dtw[r] = dt_w_c[(sc * CINNER + d) * DTRANK + r];
  float dtb = dt_b_c[sc * CINNER + d];
  float Dp  = D_c[sc * CINNER + d];
  size_t ci = (size_t)CS_SEQ_TOTAL + (((size_t)sb * CINNER + d) * NSTATE + n) * PC_CH + chunk;
  float h = phase ? cs_hinit[ci] : 0.f;
  float aprod = 1.f;
  int t0 = chunk * CS_CH;
  for (int tt = 0; tt < CS_CH; ++tt) {
    int t = t0 + tt;
    if (threadIdx.x < NPROJ) sh[threadIdx.x] = (float)xdbl[dbb + (size_t)threadIdx.x * TCHAN + t];
    __syncthreads();
    float x = (float)xconv[xcb + t];
    float pre = dtb;
#pragma unroll
    for (int r = 0; r < DTRANK; ++r) pre += dtw[r] * sh[r];
    float dt = softplusf(pre);
    float a = __expf(dt * Av);
    h = a * h + dt * sh[DTRANK + n] * x;
    if (phase) {
      float yp = h * sh[DTRANK + NSTATE + n];
      yp += __shfl_xor(yp, 8, 16);
      yp += __shfl_xor(yp, 4, 16);
      yp += __shfl_xor(yp, 2, 16);
      yp += __shfl_xor(yp, 1, 16);
      if (n == 0) {
        float z = xz[xz_elem(s, b, CINNER + d, t, TCHAN)];
        ybuf[xcb + t] = (yp + Dp * x) * siluf(z);
      }
    } else {
      aprod *= a;
    }
    __syncthreads();
  }
  if (!phase) { cs_hend[ci] = h; cs_aprod[ci] = aprod; }
}

// ------- K5c: serial chunk-combine; one thread per (scan,b,d,n) row ----------
__global__ void k_scanB(const float* __restrict__ cs_hend, const float* __restrict__ cs_aprod,
                        float* __restrict__ cs_hinit) {
  int id = blockIdx.x * blockDim.x + threadIdx.x;    // exactly 26624 launched
  size_t base; int P;
  if (id < NB * 2 * DINNER * NSTATE) { base = (size_t)id * PC_SEQ; P = PC_SEQ; }
  else { base = (size_t)CS_SEQ_TOTAL + (size_t)(id - NB * 2 * DINNER * NSTATE) * PC_CH; P = PC_CH; }
  float h = 0.f;
  for (int c = 0; c < P; ++c) {
    cs_hinit[base + c] = h;
    h = cs_aprod[base + c] * h + cs_hend[base + c];
  }
}

// ---------------- K6: deterministic gather-sum of 6 branches -> f16 ----------
__global__ void k_combine(const float* __restrict__ ybuf, _Float16* __restrict__ outall_h) {
  int id = blockIdx.x * blockDim.x + threadIdx.x;     // exactly NB*DINNER*SEQL
  int b = id / (DINNER * SEQL);
  int r = id % (DINNER * SEQL);
  int d = r / SEQL, l = r % SEQL;
  float v = ybuf[(size_t)(0 * NB + b) * PERSB + (size_t)d * SEQL + l];
  v +=      ybuf[(size_t)(1 * NB + b) * PERSB + (size_t)d * SEQL + (SEQL - 1 - l)];
  { int ci = l >> 6, j = d * 64 + (l & 63);
    v += ybuf[(size_t)(2 * NB + b) * PERSB + (size_t)ci * TCHAN + j];
    v += ybuf[(size_t)(4 * NB + b) * PERSB + (size_t)ci * TCHAN + j]; }
  { int lp = SEQL - 1 - l; int ci = lp >> 6, j = d * 64 + (lp & 63);
    v += ybuf[(size_t)(3 * NB + b) * PERSB + (size_t)ci * TCHAN + j];
    v += ybuf[(size_t)(5 * NB + b) * PERSB + (size_t)ci * TCHAN + j]; }
  outall_h[id] = (_Float16)v;
}

// ---------------- K7: out[b,l,o] = W_out(192x384) @ out_all via WMMA ---------
__global__ void k_out_proj(const _Float16* __restrict__ w_out_h,
                           const _Float16* __restrict__ outall_h,
                           float* __restrict__ out) {
  int lane = threadIdx.x & 31, wid = threadIdx.x >> 5;
  int tile = blockIdx.x * 4 + wid;             // 1536 tiles = 2 * 12 * 64
  int b = tile / 768, r = tile % 768;
  int o0 = (r / 64) * 16, l0 = (r % 64) * 16;
  bool hi = lane >= 16;
  int mr = lane & 15;
  v8f acc = {};
#pragma unroll
  for (int k0 = 0; k0 < DINNER; k0 += 32) {
    v16h a, bm;
#pragma unroll
    for (int v = 0; v < 8; ++v)
#pragma unroll
      for (int j = 0; j < 2; ++j) {
        int k = ((v < 4) ? (2 * v + j) : (16 + 2 * (v - 4) + j)) + (hi ? 8 : 0);
        a[2 * v + j] = w_out_h[(size_t)(o0 + mr) * DINNER + (k0 + k)];
      }
    int kb = hi ? 8 : 0;
#pragma unroll
    for (int i = 0; i < 16; ++i)
      bm[i] = outall_h[(size_t)b * DINNER * SEQL + (size_t)(k0 + kb + (i < 8 ? i : i + 8)) * SEQL + (l0 + mr)];
    acc = __builtin_amdgcn_wmma_f32_16x16x32_f16(false, a, false, bm, (short)0, acc, false, false);
  }
#pragma unroll
  for (int v = 0; v < 8; ++v) {
    int m = v + (hi ? 8 : 0);
    out[(size_t)b * SEQL * DMODEL + (size_t)(l0 + mr) * DMODEL + (o0 + m)] = acc[v];
  }
}

extern "C" void kernel_launch(void* const* d_in, const int* in_sizes, int n_in,
                              void* d_out, int out_size, void* d_ws, size_t ws_size,
                              hipStream_t stream) {
  (void)in_sizes; (void)n_in; (void)out_size; (void)ws_size;
  const float* hidden     = (const float*)d_in[0];
  const float* in_proj_w  = (const float*)d_in[1];
  const float* out_proj_w = (const float*)d_in[2];
  const float* conv_w_s   = (const float*)d_in[3];
  const float* conv_b_s   = (const float*)d_in[4];
  const float* xproj_w_s  = (const float*)d_in[5];
  const float* dt_w_s     = (const float*)d_in[6];
  const float* dt_b_s     = (const float*)d_in[7];
  const float* A_log_s    = (const float*)d_in[8];
  const float* D_s        = (const float*)d_in[9];
  const float* conv_w_c   = (const float*)d_in[10];
  const float* conv_b_c   = (const float*)d_in[11];
  const float* xproj_w_c  = (const float*)d_in[12];
  const float* dt_w_c     = (const float*)d_in[13];
  const float* dt_b_c     = (const float*)d_in[14];
  const float* A_log_c    = (const float*)d_in[15];
  const float* D_c        = (const float*)d_in[16];
  float* out = (float*)d_out;

  char* p = (char*)d_ws;
  auto take = [&](size_t bytes) { char* q = p; p += (bytes + 255) & ~(size_t)255; return q; };
  float*    xz       = (float*)take(sizeof(float) * (size_t)NB * E2 * SEQL);
  float*    ybuf     = (float*)take(sizeof(float) * 6ull * NB * PERSB);
  float*    cs_hend  = (float*)take(sizeof(float) * (CS_SEQ_TOTAL + CS_CH_TOTAL));
  float*    cs_aprod = (float*)take(sizeof(float) * (CS_SEQ_TOTAL + CS_CH_TOTAL));
  float*    cs_hinit = (float*)take(sizeof(float) * (CS_SEQ_TOTAL + CS_CH_TOTAL));
  _Float16* hidden_h = (_Float16*)take(2ull * NB * SEQL * DMODEL);
  _Float16* w_in_h   = (_Float16*)take(2ull * E2 * DMODEL);
  _Float16* w_out_h  = (_Float16*)take(2ull * DMODEL * DINNER);
  _Float16* xproj_h  = (_Float16*)take(2ull * 6 * NPADR * DINNER);
  _Float16* xconv    = (_Float16*)take(2ull * 6 * NB * PERSB);
  _Float16* xdbl     = (_Float16*)take(2ull * (4ull * NPADR * SEQL + 8ull * (size_t)NPADR * TCHAN));
  _Float16* outall_h = (_Float16*)take(2ull * NB * DINNER * SEQL);

  k_prep<<<2832, 256, 0, stream>>>(hidden, in_proj_w, out_proj_w, xproj_w_s, xproj_w_c,
                                   hidden_h, w_in_h, w_out_h, xproj_h);
  k_in_proj<<<1536, 128, 0, stream>>>(w_in_h, hidden_h, xz);
  k_conv_silu<<<18432, 256, 0, stream>>>(xz, conv_w_s, conv_b_s, conv_w_c, conv_b_c, xconv);
  k_xdbl<<<9408, 128, 0, stream>>>(xconv, xproj_h, xdbl);

  // phase 0: per-chunk end-state + diagonal products
  k_scan_seq2<<<192, 1024, 0, stream>>>(0, xz, xconv, xdbl, dt_w_s, dt_b_s, A_log_s, D_s,
                                        cs_hend, cs_aprod, cs_hinit, ybuf);
  k_scan_ch2<<<768, 256, 0, stream>>>(0, xz, xconv, xdbl, dt_w_c, dt_b_c, A_log_c, D_c,
                                      cs_hend, cs_aprod, cs_hinit, ybuf);
  // serial combine of chunk states (26624 independent rows)
  k_scanB<<<104, 256, 0, stream>>>(cs_hend, cs_aprod, cs_hinit);
  // phase 1: exact re-scan seeded with h_init, gated outputs
  k_scan_seq2<<<192, 1024, 0, stream>>>(1, xz, xconv, xdbl, dt_w_s, dt_b_s, A_log_s, D_s,
                                        cs_hend, cs_aprod, cs_hinit, ybuf);
  k_scan_ch2<<<768, 256, 0, stream>>>(1, xz, xconv, xdbl, dt_w_c, dt_b_c, A_log_c, D_c,
                                      cs_hend, cs_aprod, cs_hinit, ybuf);

  k_combine<<<3072, 256, 0, stream>>>(ybuf, outall_h);
  k_out_proj<<<384, 128, 0, stream>>>(w_out_h, outall_h, out);
}
